// GNOModel_37838661877948
// MI455X (gfx1250) — matrix operand order repaired and checked by reading, not compiled, not run on hardware
//
#include <hip/hip_runtime.h>

// ---------------------------------------------------------------------------
// GNO fused kernel for MI455X (gfx1250, wave32, WMMA 16x16x32 f16)
//
// Pipeline:
//   1) embed_kernel : sinusoidal embeddings of y (50000x192) and x (32768x192)
//                     stored as fp16 in workspace.
//   2) wprep_kernel : weights fp32 [K][N] -> fp16 transposed [N][K] so the
//                     WMMA B operand is one contiguous 32B load per lane.
//   3) gno_fused    : 64 edges / block, 4 waves; whole MLP
//                     384 ->128 ->256 ->128 ->32 via v_wmma_f32_16x16x32_f16
//                     with GELU epilogues staged through LDS, then the
//                     kern * f_y gather-reduce over k=8 neighbors.
// ---------------------------------------------------------------------------

typedef __attribute__((ext_vector_type(16))) _Float16 v16h;
typedef __attribute__((ext_vector_type(8)))  _Float16 v8h;
typedef __attribute__((ext_vector_type(8)))  float    v8f;

#define NUM_FREQ 32
#define EMB_DIM  192          // 3 * 32 * 2
#define KIN      384
#define N_Y      50000
#define N_X      32768
#define KNBR     8
#define C_OUT    32
#define T_DIM    4
#define TILE_E   64           // edges per block
#define S1       136          // LDS stride (halves) for 128-wide activations (+8 pad)
#define S2       264          // LDS stride (halves) for 256-wide activations (+8 pad)

// gelu(x) = 0.5*x*(1+tanh(k0*(x+0.044715 x^3)))  ==  x * sigmoid(2*k0*(x+c x^3))
// sigmoid(z) = 1/(1+exp2(-z*log2(e))).  Constants folded:
//   2*k0*log2e          = 2.30221173
//   2*k0*0.044715*log2e = 0.10294934  (2.30221173 * 0.044715)
__device__ __forceinline__ float gelu_f(float x) {
    float t = x * x;
    float w = x * fmaf(t, -0.10294934f, -2.30221173f);  // -(2 k0 log2e)(x + c x^3) / x * x
    float s = 1.0f / (1.0f + exp2f(w));                 // v_exp_f32 + rcp
    return x * s;
}

// ---------------------------------------------------------------------------
// Sinusoidal embedding:  emb[i, d*64 + 2j + {0:sin,1:cos}] = sin/cos(p[i,d]*f_j)
// f_j = 10000^(-j/32)
// ---------------------------------------------------------------------------
__global__ void embed_kernel(const float* __restrict__ p,
                             _Float16* __restrict__ emb, int n) {
    int tid = blockIdx.x * blockDim.x + threadIdx.x;   // one per (i, d, j)
    int total = n * 96;
    if (tid >= total) return;
    int i = tid / 96;
    int dj = tid - i * 96;
    int d = dj >> 5;          // 0..2
    int j = dj & 31;          // 0..31
    // 10000^(-j/32) = exp2(-j * log2(10000)/32)
    float freq = exp2f(-(float)j * (13.287712379549449f * 0.03125f));
    float ang  = p[i * 3 + d] * freq;
    float s = sinf(ang);
    float c = cosf(ang);
    _Float16* o = emb + (size_t)i * EMB_DIM + d * 64 + 2 * j;
    o[0] = (_Float16)s;
    o[1] = (_Float16)c;
}

// ---------------------------------------------------------------------------
// Weight prep: W fp32 row-major [K][N]  ->  Wt fp16 [N][K]
// ---------------------------------------------------------------------------
__global__ void wprep_kernel(const float* __restrict__ W,
                             _Float16* __restrict__ Wt, int K, int N) {
    int tid = blockIdx.x * blockDim.x + threadIdx.x;
    if (tid >= K * N) return;
    int k = tid / N;
    int n = tid - k * N;
    Wt[(size_t)n * K + k] = (_Float16)W[(size_t)k * N + n];
}

// ---------------------------------------------------------------------------
// LDS-A GEMM layer: A fp16 in LDS (per-wave 16-row strip), B = Wt[N][K] in
// global (L2/L0 resident), D -> GELU -> fp16 LDS.
// A-operand lane layout (16x32 f16): row = lane&15; lane<16 holds K 0..15,
// lane>=16 holds K 16..31 (same order as B)  -> two 16B LDS loads.
// B-operand lane layout (32x16 f16): col = lane&15; lanes<16 K 0..15,
// lanes>=16 K 16..31 -> one contiguous 32B load from Wt.
// D layout: VGPR j = (row rowBase + j + (lane>=16)*8, col nBase + lane&15).
// ---------------------------------------------------------------------------
template<int KTOT, int NTILES>
__device__ __forceinline__ void gemm_lds_f16(
    const _Float16* __restrict__ sAbase, int sa_stride, int rowBase,
    const _Float16* __restrict__ Wt, const float* __restrict__ bias,
    int nBase, _Float16* __restrict__ sOut, int so_stride, int lane) {
    const int rm    = lane & 15;
    const int koff  = (lane >> 4) << 3;   // 0 or 8  (D row half)
    const int bkoff = (lane >> 4) << 4;   // 0 or 16 (B K half)
    v8f acc[NTILES] = {};
    const _Float16* Arow = sAbase + (size_t)(rowBase + rm) * sa_stride;
#pragma unroll
    for (int kb = 0; kb < KTOT; kb += 32) {
        union { v16h v; v8h h[2]; } ua;
        ua.h[0] = *(const v8h*)(Arow + kb + bkoff);
        ua.h[1] = *(const v8h*)(Arow + kb + bkoff + 8);
        v16h a = ua.v;
#pragma unroll
        for (int t = 0; t < NTILES; ++t) {
            int n = nBase + t * 16 + rm;
            v16h b = *(const v16h*)(Wt + (size_t)n * KTOT + kb + bkoff);
            acc[t] = __builtin_amdgcn_wmma_f32_16x16x32_f16(
                false, a, false, b, (short)0, acc[t], false, false);
        }
    }
#pragma unroll
    for (int t = 0; t < NTILES; ++t) {
        int n = nBase + t * 16 + rm;
        float bv = bias[n];
#pragma unroll
        for (int j = 0; j < 8; ++j) {
            int m = rowBase + j + koff;
            sOut[(size_t)m * so_stride + n] = (_Float16)gelu_f(acc[t][j] + bv);
        }
    }
}

// Layer 0: A rows gathered from global embeddings (cols 0..191 = y_emb[idx[e]],
// cols 192..383 = x_emb[e>>3]); K panels of 32 never straddle the boundary.
template<int NTILES>
__device__ __forceinline__ void gemm_l0(
    const _Float16* __restrict__ y_emb, const _Float16* __restrict__ x_emb,
    const int* __restrict__ idxS, int eBase, int rowBase,
    const _Float16* __restrict__ Wt, const float* __restrict__ bias,
    _Float16* __restrict__ sOut, int so_stride, int lane) {
    const int rm    = lane & 15;
    const int koff  = (lane >> 4) << 3;
    const int bkoff = (lane >> 4) << 4;
    const int eloc  = rowBase + rm;
    const _Float16* yrow = y_emb + (size_t)idxS[eloc] * EMB_DIM;
    const _Float16* xrow = x_emb + (size_t)((eBase + eloc) >> 3) * EMB_DIM;
    v8f acc[NTILES] = {};
#pragma unroll
    for (int kb = 0; kb < KIN; kb += 32) {
        const _Float16* src = (kb < EMB_DIM) ? (yrow + kb) : (xrow + (kb - EMB_DIM));
        union { v16h v; v8h h[2]; } ua;
        ua.h[0] = *(const v8h*)(src + bkoff);
        ua.h[1] = *(const v8h*)(src + bkoff + 8);
        v16h a = ua.v;
#pragma unroll
        for (int t = 0; t < NTILES; ++t) {
            int n = t * 16 + rm;
            v16h b = *(const v16h*)(Wt + (size_t)n * KIN + kb + bkoff);
            acc[t] = __builtin_amdgcn_wmma_f32_16x16x32_f16(
                false, a, false, b, (short)0, acc[t], false, false);
        }
    }
#pragma unroll
    for (int t = 0; t < NTILES; ++t) {
        int n = t * 16 + rm;
        float bv = bias[n];
#pragma unroll
        for (int j = 0; j < 8; ++j) {
            int m = rowBase + j + koff;
            sOut[(size_t)m * so_stride + n] = (_Float16)gelu_f(acc[t][j] + bv);
        }
    }
}

// Layer 3: K=128, N=32 (2 tiles), fp32 output, bias only (no GELU).
__device__ __forceinline__ void gemm_l3(
    const _Float16* __restrict__ sAbase, int sa_stride, int rowBase,
    const _Float16* __restrict__ Wt, const float* __restrict__ bias,
    float* __restrict__ kernS, int lane) {
    const int rm    = lane & 15;
    const int koff  = (lane >> 4) << 3;
    const int bkoff = (lane >> 4) << 4;
    v8f acc[2] = {};
    const _Float16* Arow = sAbase + (size_t)(rowBase + rm) * sa_stride;
#pragma unroll
    for (int kb = 0; kb < 128; kb += 32) {
        union { v16h v; v8h h[2]; } ua;
        ua.h[0] = *(const v8h*)(Arow + kb + bkoff);
        ua.h[1] = *(const v8h*)(Arow + kb + bkoff + 8);
        v16h a = ua.v;
#pragma unroll
        for (int t = 0; t < 2; ++t) {
            int n = t * 16 + rm;
            v16h b = *(const v16h*)(Wt + (size_t)n * 128 + kb + bkoff);
            acc[t] = __builtin_amdgcn_wmma_f32_16x16x32_f16(
                false, a, false, b, (short)0, acc[t], false, false);
        }
    }
#pragma unroll
    for (int t = 0; t < 2; ++t) {
        int n = t * 16 + rm;
        float bv = bias[n];
#pragma unroll
        for (int j = 0; j < 8; ++j) {
            int m = rowBase + j + koff;
            kernS[m * C_OUT + n] = acc[t][j] + bv;
        }
    }
}

// ---------------------------------------------------------------------------
// Fused MLP + gather-reduce. 128 threads = 4 waves; 64 edges per block.
// ---------------------------------------------------------------------------
__global__ __launch_bounds__(128) void gno_fused(
    const _Float16* __restrict__ y_emb, const _Float16* __restrict__ x_emb,
    const int* __restrict__ nbr, const float* __restrict__ f_y,
    const _Float16* __restrict__ Wt0, const float* __restrict__ b0,
    const _Float16* __restrict__ Wt1, const float* __restrict__ b1,
    const _Float16* __restrict__ Wt2, const float* __restrict__ b2,
    const _Float16* __restrict__ Wt3, const float* __restrict__ b3,
    float* __restrict__ out) {
    __shared__ __align__(16) _Float16 sA[TILE_E * S1];   // H1, then H3 (17408 B)
    __shared__ __align__(16) _Float16 sB[TILE_E * S2];   // H2, then kern (33792 B)
    __shared__ int idxS[TILE_E];

    const int tid   = threadIdx.x;
    const int wave  = tid >> 5;
    const int lane  = tid & 31;
    const int eBase = blockIdx.x * TILE_E;

    if (tid < TILE_E) idxS[tid] = nbr[eBase + tid];
    __syncthreads();

    const int rowBase = wave * 16;

    // L0: [64x384] @ [384x128] -> sA
    gemm_l0<8>(y_emb, x_emb, idxS, eBase, rowBase, Wt0, b0, sA, S1, lane);
    __syncthreads();
    // L1: [64x128] @ [128x256] -> sB (two 128-col halves to cap accumulators)
    gemm_lds_f16<128, 8>(sA, S1, rowBase, Wt1, b1, 0,   sB, S2, lane);
    gemm_lds_f16<128, 8>(sA, S1, rowBase, Wt1, b1, 128, sB, S2, lane);
    __syncthreads();
    // L2: [64x256] @ [256x128] -> sA (reused)
    gemm_lds_f16<256, 8>(sB, S2, rowBase, Wt2, b2, 0, sA, S1, lane);
    __syncthreads();
    // L3: [64x128] @ [128x32] -> kern fp32 (reuse sB space)
    float* kernS = (float*)sB;
    gemm_l3(sA, S1, rowBase, Wt3, b3, kernS, lane);
    __syncthreads();

    // Final: out[t, x, c] = sum_k kern[e] * f_y[t, idx[e], c];  8 x-points/wave-row
    const int c   = tid & 31;
    const int xw  = tid >> 5;        // 0..3
#pragma unroll
    for (int xi = 0; xi < 2; ++xi) {
        int xl = xw + xi * 4;        // 0..7 local x index
        int e0 = xl * KNBR;
        float a0 = 0.f, a1 = 0.f, a2 = 0.f, a3 = 0.f;
#pragma unroll
        for (int k = 0; k < KNBR; ++k) {
            int el = e0 + k;
            float kv = kernS[el * C_OUT + c];
            size_t yr = (size_t)idxS[el] * C_OUT + c;
            a0 += kv * f_y[yr];
            a1 += kv * f_y[(size_t)1 * N_Y * C_OUT + yr];
            a2 += kv * f_y[(size_t)2 * N_Y * C_OUT + yr];
            a3 += kv * f_y[(size_t)3 * N_Y * C_OUT + yr];
        }
        size_t xg = (size_t)(eBase / KNBR + xl) * C_OUT + c;
        out[xg] = a0;
        out[(size_t)1 * N_X * C_OUT + xg] = a1;
        out[(size_t)2 * N_X * C_OUT + xg] = a2;
        out[(size_t)3 * N_X * C_OUT + xg] = a3;
    }
}

// ---------------------------------------------------------------------------
extern "C" void kernel_launch(void* const* d_in, const int* in_sizes, int n_in,
                              void* d_out, int out_size, void* d_ws, size_t ws_size,
                              hipStream_t stream) {
    const float* y   = (const float*)d_in[0];
    const float* x   = (const float*)d_in[1];
    const float* f_y = (const float*)d_in[2];
    const int*   nbr = (const int*)d_in[3];
    const float* W0 = (const float*)d_in[4];  const float* b0 = (const float*)d_in[5];
    const float* W1 = (const float*)d_in[6];  const float* b1 = (const float*)d_in[7];
    const float* W2 = (const float*)d_in[8];  const float* b2 = (const float*)d_in[9];
    const float* W3 = (const float*)d_in[10]; const float* b3 = (const float*)d_in[11];
    float* out = (float*)d_out;

    // Workspace carve (256B-aligned so all v16h/v8h loads are 32B-aligned)
    char* ws = (char*)d_ws;
    size_t off = 0;
    auto alloc = [&](size_t bytes) -> char* {
        char* p = ws + off;
        off += (bytes + 255) & ~(size_t)255;
        return p;
    };
    _Float16* y_emb = (_Float16*)alloc((size_t)N_Y * EMB_DIM * 2);   // 19.2 MB
    _Float16* x_emb = (_Float16*)alloc((size_t)N_X * EMB_DIM * 2);   // 12.6 MB
    _Float16* Wt0   = (_Float16*)alloc((size_t)KIN * 128 * 2);
    _Float16* Wt1   = (_Float16*)alloc((size_t)128 * 256 * 2);
    _Float16* Wt2   = (_Float16*)alloc((size_t)256 * 128 * 2);
    _Float16* Wt3   = (_Float16*)alloc((size_t)128 * 32 * 2);

    // 1) embeddings
    embed_kernel<<<(N_Y * 96 + 255) / 256, 256, 0, stream>>>(y, y_emb, N_Y);
    embed_kernel<<<(N_X * 96 + 255) / 256, 256, 0, stream>>>(x, x_emb, N_X);

    // 2) weight transpose + fp16 convert
    wprep_kernel<<<(KIN * 128 + 255) / 256, 256, 0, stream>>>(W0, Wt0, KIN, 128);
    wprep_kernel<<<(128 * 256 + 255) / 256, 256, 0, stream>>>(W1, Wt1, 128, 256);
    wprep_kernel<<<(256 * 128 + 255) / 256, 256, 0, stream>>>(W2, Wt2, 256, 128);
    wprep_kernel<<<(128 * 32 + 255) / 256, 256, 0, stream>>>(W3, Wt3, 128, 32);

    // 3) fused MLP + reduce: 262144 edges / 64 per block
    gno_fused<<<(N_X * KNBR) / TILE_E, 128, 0, stream>>>(
        y_emb, x_emb, nbr, f_y, Wt0, b0, Wt1, b1, Wt2, b2, Wt3, b3, out);
}
